// RGCNNodeClassifier_1726576853586
// MI455X (gfx1250) — compile-verified, matrix operand order checked by reading, not run on hardware
//
#include <hip/hip_runtime.h>
#include <hip/hip_bf16.h>

// ---------------------------------------------------------------------------
// RGCN node classifier for MI455X (gfx1250, wave32, WMMA).
// Dense transforms via v_wmma_f32_16x16x32_bf16 (fp32 accumulate),
// edge aggregation via global_atomic_add_f32, mean via precomputed 1/deg.
// ---------------------------------------------------------------------------

typedef __attribute__((ext_vector_type(16))) __bf16 v16bf;
typedef __attribute__((ext_vector_type(8)))  float  v8f;

#define KDIM 128   // in/hidden channels (both layers are 128 -> 128)
#define TM   64    // rows of C per workgroup
#define PAD  132   // LDS row stride in bf16 elems (keeps pairs 4B-aligned, kills bank conflicts)

// C[M x 128] = A[M x 128] @ B[128 x 128] (+ bias). A,B,C fp32 row-major in memory;
// A and B are converted to bf16 in LDS and fed to WMMA with fp32 accumulators.
__global__ __launch_bounds__(256)
void rgcn_gemm_bf16(const float* __restrict__ A, const float* __restrict__ B,
                    const float* __restrict__ bias, float* __restrict__ C, int M)
{
    __shared__ __bf16 lB[KDIM * PAD];  // staged transposed: lB[n][k]
    __shared__ __bf16 lA[TM   * PAD];  // lA[r][k]

    const int tid     = threadIdx.x;
    const int rowBase = blockIdx.x * TM;

    // Stage weights (128x128 fp32 -> bf16, transposed so K is contiguous per column n)
    for (int i = tid; i < KDIM * KDIM; i += 256) {
        int k = i >> 7, n = i & 127;
        lB[n * PAD + k] = (__bf16)B[k * KDIM + n];
    }
    // Stage A tile (64 rows), zero-pad the tail rows
    for (int i = tid; i < TM * KDIM; i += 256) {
        int r = i >> 7, k = i & 127;
        int row = rowBase + r;
        lA[r * PAD + k] = (__bf16)((row < M) ? A[(size_t)row * KDIM + k] : 0.0f);
    }
    __syncthreads();

    const int lane    = tid & 31;
    const int wid     = tid >> 5;
    const int rowTile = (wid & 3) * 16;     // 4 row-stripes of 16
    const int colBase = (wid >> 2) * 64;    // 2 column halves of 64
    const int mloc    = lane & 15;
    const int half    = lane >> 4;

    v8f acc[4] = {};   // 4 tiles of 16x16 f32

    for (int kk = 0; kk < KDIM; kk += 32) {
        // A fragment: 16x32 bf16  (ISA 7.12.2: lane%16 = M; VGPR v holds K pair)
        v16bf a;
        const __bf16* ap = &lA[(rowTile + mloc) * PAD];
#pragma unroll
        for (int v = 0; v < 8; ++v) {
            int kb = kk + ((v < 4) ? (2 * v) : (16 + 2 * (v - 4))) + 8 * half;
            a[2 * v]     = ap[kb];
            a[2 * v + 1] = ap[kb + 1];
        }
#pragma unroll
        for (int t = 0; t < 4; ++t) {
            // B fragment: 32x16 bf16, mirrored layout (lane%16 = N), K contiguous via transpose
            const __bf16* bp = &lB[(colBase + t * 16 + mloc) * PAD];
            v16bf b;
#pragma unroll
            for (int v = 0; v < 8; ++v) {
                int kb = kk + ((v < 4) ? (2 * v) : (16 + 2 * (v - 4))) + 8 * half;
                b[2 * v]     = bp[kb];
                b[2 * v + 1] = bp[kb + 1];
            }
            acc[t] = __builtin_amdgcn_wmma_f32_16x16x32_bf16(
                false, a, false, b, (short)0, acc[t], false, false);
        }
    }

    // Epilogue: C layout per ISA (VGPR g: M = half*8 + g, lane%16 = N)
#pragma unroll
    for (int t = 0; t < 4; ++t) {
        int col  = colBase + t * 16 + mloc;
        float bb = bias ? bias[col] : 0.0f;
#pragma unroll
        for (int g = 0; g < 8; ++g) {
            int row = rowBase + rowTile + half * 8 + g;
            if (row < M) C[(size_t)row * KDIM + col] = acc[t][g] + bb;
        }
    }
}

// ------------------------- aggregation helpers ------------------------------

__global__ __launch_bounds__(256)
void rgcn_fill_zero(float* __restrict__ p, int n)
{
    int i = blockIdx.x * 256 + threadIdx.x;
    if (i < n) p[i] = 0.0f;
}

__global__ __launch_bounds__(256)
void rgcn_count(const int* __restrict__ dst, const int* __restrict__ et,
                float* __restrict__ cnt, int N, int E)
{
    int e = blockIdx.x * 256 + threadIdx.x;
    if (e < E) atomicAdd(&cnt[(size_t)et[e] * N + dst[e]], 1.0f);
}

__global__ __launch_bounds__(256)
void rgcn_recip(float* __restrict__ cnt, int n)
{
    int i = blockIdx.x * 256 + threadIdx.x;
    if (i < n) cnt[i] = 1.0f / fmaxf(cnt[i], 1.0f);
}

// One wave per edge; each lane scatters 4 channels (128 = 32 lanes * float4).
__global__ __launch_bounds__(256)
void rgcn_scatter(const float* __restrict__ XR, const int* __restrict__ src,
                  const int* __restrict__ dst, const int* __restrict__ et,
                  int rel, const float* __restrict__ rinv,
                  float* __restrict__ out, int E)
{
    int e = blockIdx.x * 8 + (threadIdx.x >> 5);
    if (e >= E) return;
    if (et[e] != rel) return;
    int lane = threadIdx.x & 31;
    int s = src[e], d = dst[e];
    float w = rinv[d];
    float4 v = ((const float4*)(XR + (size_t)s * KDIM))[lane];
    float* o = out + (size_t)d * KDIM + lane * 4;
    atomicAdd(o + 0, v.x * w);
    atomicAdd(o + 1, v.y * w);
    atomicAdd(o + 2, v.z * w);
    atomicAdd(o + 3, v.w * w);
}

__global__ __launch_bounds__(256)
void rgcn_relu(float* __restrict__ p, int n)
{
    int i = blockIdx.x * 256 + threadIdx.x;
    if (i < n) p[i] = fmaxf(p[i], 0.0f);
}

// One wave per node: dot(h[node], Wc) + bc with a wave32 shuffle reduction.
__global__ __launch_bounds__(256)
void rgcn_classify(const float* __restrict__ h, const float* __restrict__ Wc,
                   const float* __restrict__ bc, float* __restrict__ out, int M)
{
    int node = blockIdx.x * 8 + (threadIdx.x >> 5);
    if (node >= M) return;
    int lane = threadIdx.x & 31;
    float4 hv = ((const float4*)(h + (size_t)node * KDIM))[lane];
    float4 wv = ((const float4*)Wc)[lane];
    float p = hv.x * wv.x + hv.y * wv.y + hv.z * wv.z + hv.w * wv.w;
#pragma unroll
    for (int off = 16; off > 0; off >>= 1) p += __shfl_down(p, off, 32);
    if (lane == 0) out[node] = p + bc[0];
}

// ---------------------------------------------------------------------------

extern "C" void kernel_launch(void* const* d_in, const int* in_sizes, int n_in,
                              void* d_out, int out_size, void* d_ws, size_t ws_size,
                              hipStream_t stream)
{
    (void)n_in; (void)out_size; (void)ws_size;

    const float* x     = (const float*)d_in[0];
    const int*   eidx  = (const int*)d_in[1];
    const int*   etype = (const int*)d_in[2];
    const float* W1    = (const float*)d_in[3];
    const float* root1 = (const float*)d_in[4];
    const float* b1    = (const float*)d_in[5];
    const float* W2    = (const float*)d_in[6];
    const float* root2 = (const float*)d_in[7];
    const float* b2    = (const float*)d_in[8];
    const float* Wc    = (const float*)d_in[9];
    const float* bc    = (const float*)d_in[10];

    const int N = in_sizes[0] / KDIM;
    const int E = in_sizes[2];
    const int R = in_sizes[3] / (KDIM * KDIM);

    const int* src = eidx;
    const int* dst = eidx + E;

    char*  ws    = (char*)d_ws;
    size_t featB = (size_t)N * KDIM * sizeof(float);
    float* bufH1 = (float*)(ws);
    float* bufH2 = (float*)(ws + featB);
    float* bufXR = (float*)(ws + 2 * featB);
    float* cnt   = (float*)(ws + 3 * featB);   // R*N floats: counts -> reciprocals

    const int cn       = R * N;
    const int gemmGrid = (N + TM - 1) / TM;
    const int edgeGrid = (E + 7) / 8;
    const int featGrid = ((int)((size_t)N * KDIM) + 255) / 256;

    // Per-(relation,dst) in-degree -> 1/max(deg,1), shared by both layers.
    rgcn_fill_zero<<<(cn + 255) / 256, 256, 0, stream>>>(cnt, cn);
    rgcn_count<<<(E + 255) / 256, 256, 0, stream>>>(dst, etype, cnt, N, E);
    rgcn_recip<<<(cn + 255) / 256, 256, 0, stream>>>(cnt, cn);

    // ---- Layer 1: H1 = relu(x@root1 + b1 + sum_r mean_r(x@W1[r])) ----
    rgcn_gemm_bf16<<<gemmGrid, 256, 0, stream>>>(x, root1, b1, bufH1, N);
    for (int r = 0; r < R; ++r) {
        rgcn_gemm_bf16<<<gemmGrid, 256, 0, stream>>>(
            x, W1 + (size_t)r * KDIM * KDIM, nullptr, bufXR, N);
        rgcn_scatter<<<edgeGrid, 256, 0, stream>>>(
            bufXR, src, dst, etype, r, cnt + (size_t)r * N, bufH1, E);
    }
    rgcn_relu<<<featGrid, 256, 0, stream>>>(bufH1, N * KDIM);

    // ---- Layer 2: H2 = relu(H1@root2 + b2 + sum_r mean_r(H1@W2[r])) ----
    rgcn_gemm_bf16<<<gemmGrid, 256, 0, stream>>>(bufH1, root2, b2, bufH2, N);
    for (int r = 0; r < R; ++r) {
        rgcn_gemm_bf16<<<gemmGrid, 256, 0, stream>>>(
            bufH1, W2 + (size_t)r * KDIM * KDIM, nullptr, bufXR, N);
        rgcn_scatter<<<edgeGrid, 256, 0, stream>>>(
            bufXR, src, dst, etype, r, cnt + (size_t)r * N, bufH2, E);
    }
    rgcn_relu<<<featGrid, 256, 0, stream>>>(bufH2, N * KDIM);

    // ---- Classifier: out[i] = dot(H2[i], Wc) + bc ----
    rgcn_classify<<<(N + 7) / 8, 256, 0, stream>>>(bufH2, Wc, bc, (float*)d_out, N);
}
